// ColBERTModel_45921790329433
// MI455X (gfx1250) — compile-verified, matrix operand order checked by reading, not run on hardware
//
#include <hip/hip_runtime.h>
#include <hip/hip_bf16.h>
#include <cstdint>
#include <cstddef>

// ---------------------------------------------------------------------------
// ColBERT scoring on MI455X (gfx1250, wave32, WMMA).
// Phase 1: E = normalize(embed[ids] @ W + b) in bf16 via v_wmma_f32_16x16x32_bf16
// Phase 2: scores[b] = sum_q qmask * max_t(masked <Qe, De>) via the same WMMA.
// ---------------------------------------------------------------------------

typedef __attribute__((ext_vector_type(16))) __bf16   v16bf;
typedef __attribute__((ext_vector_type(8)))  float    v8f;
typedef __attribute__((ext_vector_type(8)))  unsigned v8u;

#define VOCAB   30522
#define HIDDEN  768
#define EMB     128
#define NBATCH  1024
#define QLEN    32
#define DLEN    180
#define NEGV    (-1.0e4f)

// pack two f32 into one u32 of bf16 pair (lo -> [15:0], hi -> [31:16]), RNE
__device__ __forceinline__ unsigned pack_bf16(float lo, float hi) {
  unsigned a = __builtin_bit_cast(unsigned, lo);
  unsigned b = __builtin_bit_cast(unsigned, hi);
  a += 0x7FFFu + ((a >> 16) & 1u);
  b += 0x7FFFu + ((b >> 16) & 1u);
  return (a >> 16) | (b & 0xFFFF0000u);
}

__device__ __forceinline__ unsigned short f32_to_bf16u(float v) {
  unsigned u = __builtin_bit_cast(unsigned, v);
  u += 0x7FFFu + ((u >> 16) & 1u);
  return (unsigned short)(u >> 16);
}

// 16-bit A-fragment K offset for (vgpr j, lane half): j0..3 -> 2j, j4..7 -> 16+2(j-4),
// upper half lanes are +8 (per ISA 16-bit 16x32 A layout; B mirrors with M->N).
__device__ __forceinline__ int frag_koff(int j, int half) {
  return (j < 4 ? 2 * j : 16 + 2 * (j - 4)) + 8 * half;
}

// ---------------------------------------------------------------------------
// Phase 1: gather + projection GEMM + bias + L2 normalize, output bf16.
// Block = 256 threads (8 waves). LDS = full proj_w as bf16 fragments (192 KB).
// Each wave processes 16-token tiles: M=16, N=128 (8 tiles), K=768 (24 steps).
// ---------------------------------------------------------------------------
__global__ __launch_bounds__(256) void colbert_encode_kernel(
    const int* __restrict__ ids, int ntiles,
    const float* __restrict__ embed, const float* __restrict__ W,
    const float* __restrict__ bias, unsigned short* __restrict__ Eout)
{
  extern __shared__ unsigned Wlds[];   // [24 ksteps][8 ntiles][lane*8 + j] = 49152 u32
  const int tid = threadIdx.x;

  // Stage proj_w into LDS, pre-swizzled into B-fragment layout (one time).
  for (int idx = tid; idx < 24 * 8 * 256; idx += 256) {
    const int ks = idx >> 11;          // k-step
    const int nt = (idx >> 8) & 7;     // n-tile
    const int e  = idx & 255;
    const int l  = e >> 3;             // lane
    const int j  = e & 7;              // vgpr slot
    const int half = l >> 4;
    const int n  = l & 15;
    const int k  = ks * 32 + frag_koff(j, half);
    const int col = nt * 16 + n;
    Wlds[idx] = pack_bf16(W[k * EMB + col], W[(k + 1) * EMB + col]);
  }
  __syncthreads();

  const int lane = tid & 31;
  const int wv   = tid >> 5;
  const int half = lane >> 4;
  const int lm   = lane & 15;
  const int gw   = blockIdx.x * 8 + wv;
  const int nw   = gridDim.x * 8;

  float bcol[8];
  #pragma unroll
  for (int t = 0; t < 8; ++t) bcol[t] = bias[t * 16 + lm];

  for (int tile = gw; tile < ntiles; tile += nw) {
    const int id = ids[tile * 16 + lm];                       // row for M = lm (both halves)
    const float4* rp = (const float4*)(embed + (size_t)id * HIDDEN);

    v8f acc[8] = {};                                           // 8 n-tiles of 16x16 f32 C

    for (int ks = 0; ks < 24; ++ks) {
      const int kk4 = ks * 8;                                  // k offset / 4
      // A fragment: lane holds row M=lm, K = kk + {0..7,16..23} (+8 for upper half)
      const float4 v0 = rp[kk4 + 2 * half];
      const float4 v1 = rp[kk4 + 2 * half + 1];
      const float4 v2 = rp[kk4 + 4 + 2 * half];
      const float4 v3 = rp[kk4 + 4 + 2 * half + 1];
      v8u ap;
      ap[0] = pack_bf16(v0.x, v0.y);  ap[1] = pack_bf16(v0.z, v0.w);
      ap[2] = pack_bf16(v1.x, v1.y);  ap[3] = pack_bf16(v1.z, v1.w);
      ap[4] = pack_bf16(v2.x, v2.y);  ap[5] = pack_bf16(v2.z, v2.w);
      ap[6] = pack_bf16(v3.x, v3.y);  ap[7] = pack_bf16(v3.z, v3.w);
      const v16bf av = __builtin_bit_cast(v16bf, ap);

      #pragma unroll
      for (int nt = 0; nt < 8; ++nt) {
        const uint4* lp = (const uint4*)&Wlds[(ks * 8 + nt) * 256 + lane * 8];
        const uint4 b0 = lp[0];
        const uint4 b1 = lp[1];
        v8u bp;
        bp[0] = b0.x; bp[1] = b0.y; bp[2] = b0.z; bp[3] = b0.w;
        bp[4] = b1.x; bp[5] = b1.y; bp[6] = b1.z; bp[7] = b1.w;
        const v16bf bv = __builtin_bit_cast(v16bf, bp);
        acc[nt] = __builtin_amdgcn_wmma_f32_16x16x32_bf16(
            false, av, false, bv, (short)0, acc[nt], false, false);
      }
    }

    // Epilogue: bias, row L2 norm (C layout: vgpr i -> row i+8*half, lane -> col)
    #pragma unroll
    for (int t = 0; t < 8; ++t) {
      #pragma unroll
      for (int i = 0; i < 8; ++i) acc[t][i] += bcol[t];
    }

    #pragma unroll
    for (int i = 0; i < 8; ++i) {
      float s = 0.0f;
      #pragma unroll
      for (int t = 0; t < 8; ++t) s += acc[t][i] * acc[t][i];
      s += __shfl_xor(s, 1, 32);
      s += __shfl_xor(s, 2, 32);
      s += __shfl_xor(s, 4, 32);
      s += __shfl_xor(s, 8, 32);                 // sum over 16 lanes in this half
      const float scale = 1.0f / (sqrtf(s) + 1e-12f);
      const size_t row = (size_t)tile * 16 + i + 8 * half;
      #pragma unroll
      for (int t = 0; t < 8; ++t)
        Eout[row * EMB + t * 16 + lm] = f32_to_bf16u(acc[t][i] * scale);
    }
  }
}

// ---------------------------------------------------------------------------
// Phase 2: per-batch MaxSim. One wave per batch.
// S = Qe(32x128) @ De^T(128x180) via 2 M-tiles x 12 N-tiles x 4 k-steps WMMA.
// ---------------------------------------------------------------------------
__global__ __launch_bounds__(32) void colbert_maxsim_kernel(
    const unsigned short* __restrict__ Qe, const unsigned short* __restrict__ De,
    const int* __restrict__ qmask, const int* __restrict__ dmask,
    float* __restrict__ scores)
{
  const int b    = blockIdx.x;
  const int lane = threadIdx.x & 31;
  const int half = lane >> 4;
  const int lm   = lane & 15;

  const unsigned* Qb = (const unsigned*)(Qe + (size_t)b * QLEN * EMB);
  const unsigned* Db = (const unsigned*)(De + (size_t)b * DLEN * EMB);

  // Preload Q fragments: 2 M-tiles x 4 k-steps (fragment K-pairs are contiguous u32s)
  v16bf avq[2][4];
  #pragma unroll
  for (int mt = 0; mt < 2; ++mt) {
    const uint4* qp = (const uint4*)(Qb + (size_t)(mt * 16 + lm) * (EMB / 2));
    #pragma unroll
    for (int ks = 0; ks < 4; ++ks) {
      const uint4 a0 = qp[4 * ks + half];
      const uint4 a1 = qp[4 * ks + half + 2];
      v8u ap;
      ap[0] = a0.x; ap[1] = a0.y; ap[2] = a0.z; ap[3] = a0.w;
      ap[4] = a1.x; ap[5] = a1.y; ap[6] = a1.z; ap[7] = a1.w;
      avq[mt][ks] = __builtin_bit_cast(v16bf, ap);
    }
  }

  v8f rm0, rm1;
  #pragma unroll
  for (int i = 0; i < 8; ++i) { rm0[i] = NEGV; rm1[i] = NEGV; }

  const uint4 zero4 = make_uint4(0u, 0u, 0u, 0u);

  #pragma unroll 1
  for (int nt = 0; nt < 12; ++nt) {                 // 12*16 = 192 covers DLEN=180
    const int  dtok  = nt * 16 + lm;                // doc token for this lane (B n-index)
    const bool valid = (dtok < DLEN);
    const uint4* dp  = (const uint4*)(Db + (size_t)(valid ? dtok : 0) * (EMB / 2));

    v8f a0 = {}, a1 = {};
    #pragma unroll
    for (int ks = 0; ks < 4; ++ks) {
      uint4 b0 = dp[4 * ks + half];
      uint4 b1 = dp[4 * ks + half + 2];
      b0 = valid ? b0 : zero4;
      b1 = valid ? b1 : zero4;
      v8u bp;
      bp[0] = b0.x; bp[1] = b0.y; bp[2] = b0.z; bp[3] = b0.w;
      bp[4] = b1.x; bp[5] = b1.y; bp[6] = b1.z; bp[7] = b1.w;
      const v16bf bv = __builtin_bit_cast(v16bf, bp);
      a0 = __builtin_amdgcn_wmma_f32_16x16x32_bf16(false, avq[0][ks], false, bv,
                                                   (short)0, a0, false, false);
      a1 = __builtin_amdgcn_wmma_f32_16x16x32_bf16(false, avq[1][ks], false, bv,
                                                   (short)0, a1, false, false);
    }

    const bool keep = valid && (dmask[b * DLEN + (valid ? dtok : 0)] > 0);
    #pragma unroll
    for (int i = 0; i < 8; ++i) {
      rm0[i] = fmaxf(rm0[i], keep ? a0[i] : NEGV);
      rm1[i] = fmaxf(rm1[i], keep ? a1[i] : NEGV);
    }
  }

  // Row-max across the 16 lanes of each half (covers all 192 padded columns)
  #pragma unroll
  for (int i = 0; i < 8; ++i) {
    rm0[i] = fmaxf(rm0[i], __shfl_xor(rm0[i], 1, 32));
    rm0[i] = fmaxf(rm0[i], __shfl_xor(rm0[i], 2, 32));
    rm0[i] = fmaxf(rm0[i], __shfl_xor(rm0[i], 4, 32));
    rm0[i] = fmaxf(rm0[i], __shfl_xor(rm0[i], 8, 32));
    rm1[i] = fmaxf(rm1[i], __shfl_xor(rm1[i], 1, 32));
    rm1[i] = fmaxf(rm1[i], __shfl_xor(rm1[i], 2, 32));
    rm1[i] = fmaxf(rm1[i], __shfl_xor(rm1[i], 4, 32));
    rm1[i] = fmaxf(rm1[i], __shfl_xor(rm1[i], 8, 32));
  }

  // Weighted sum over q rows: half h holds rows {i+8h} and {16+i+8h}
  float partial = 0.0f;
  #pragma unroll
  for (int i = 0; i < 8; ++i) {
    partial += rm0[i] * (float)qmask[b * QLEN + i + 8 * half];
    partial += rm1[i] * (float)qmask[b * QLEN + 16 + i + 8 * half];
  }
  const float total = partial + __shfl_xor(partial, 16, 32);
  if (lane == 0) scores[b] = total;
}

// ---------------------------------------------------------------------------
extern "C" void kernel_launch(void* const* d_in, const int* in_sizes, int n_in,
                              void* d_out, int out_size, void* d_ws, size_t ws_size,
                              hipStream_t stream) {
  (void)in_sizes; (void)n_in; (void)out_size; (void)ws_size;
  const int*   q_ids  = (const int*)d_in[0];
  const int*   q_mask = (const int*)d_in[1];
  const int*   d_ids  = (const int*)d_in[2];
  const int*   d_mask = (const int*)d_in[3];
  const float* embed  = (const float*)d_in[4];
  const float* W      = (const float*)d_in[5];
  const float* bias   = (const float*)d_in[6];
  float*       scores = (float*)d_out;

  unsigned short* Qe = (unsigned short*)d_ws;                       // 1024*32*128 bf16
  unsigned short* Dd = Qe + (size_t)NBATCH * QLEN * EMB;            // 1024*180*128 bf16

  const size_t ldsW = (size_t)24 * 8 * 256 * 4;                     // 192 KB (WGP has 320 KB)

  colbert_encode_kernel<<<256, 256, ldsW, stream>>>(
      q_ids, NBATCH * QLEN / 16, embed, W, bias, Qe);
  colbert_encode_kernel<<<256, 256, ldsW, stream>>>(
      d_ids, NBATCH * DLEN / 16, embed, W, bias, Dd);
  colbert_maxsim_kernel<<<NBATCH, 32, 0, stream>>>(
      Qe, Dd, q_mask, d_mask, scores);
}
